// E59CellPython_65189013618950
// MI455X (gfx1250) — matrix operand order repaired and checked by reading, not compiled, not run on hardware
//
#include <hip/hip_runtime.h>
#include <stdint.h>

// ---------------------------------------------------------------------------
// E59 cell: out = h*silu(h), h_t = rmsnorm(h_{t-1} + alpha * (x_t @ W^T + b))
// Phase 0: convert W -> split bf16 hi/lo (stashed in out[] region, 4MB)
// Phase 1: split-bf16 WMMA GEMM  Wx = X @ W^T + b  -> stored into h[1..T]
//          (B tiles moved global->LDS with async copies when available)
// Phase 2: per-row wave32 latency-optimized sequential scan (in-place on h)
// ---------------------------------------------------------------------------

typedef __attribute__((ext_vector_type(16))) __bf16 v16bf;
typedef __attribute__((ext_vector_type(8)))  float  v8f;
typedef __attribute__((ext_vector_type(4)))  int    v4i;

#if defined(__has_builtin)
#if __has_builtin(__builtin_amdgcn_global_load_async_to_lds_b128)
#define HAVE_ASYNC_LDS 1
#endif
#endif
#ifndef HAVE_ASYNC_LDS
#define HAVE_ASYNC_LDS 0
#endif

// async copy param types (from clang diagnostic): (AS1 v4i*, AS3 v4i*, imm, imm)
#define ASYNC_GSRC(p) ((__attribute__((address_space(1))) v4i*)(p))
#define ASYNC_LDST(p) ((__attribute__((address_space(3))) v4i*)(p))

__device__ __forceinline__ void wait_async0() {
#if defined(__has_builtin) && __has_builtin(__builtin_amdgcn_s_wait_asynccnt)
  __builtin_amdgcn_s_wait_asynccnt(0);
#else
  asm volatile("s_wait_asynccnt 0x0" ::: "memory");
#endif
}

// Split one f32 into bf16 hi + bf16 lo (residual), 8 at a time, packed so that
// consecutive k-pairs land in consecutive VGPR halves (matches WMMA layouts).
__device__ __forceinline__ void split8(const float4 a, const float4 b,
                                       uint4& hi, uint4& lo) {
  float f[8] = {a.x, a.y, a.z, a.w, b.x, b.y, b.z, b.w};
  union { uint4 q; __bf16 e[8]; } H, L;
#pragma unroll
  for (int i = 0; i < 8; ++i) {
    __bf16 h = (__bf16)f[i];              // RNE f32->bf16
    H.e[i] = h;
    L.e[i] = (__bf16)(f[i] - (float)h);   // residual
  }
  hi = H.q; lo = L.q;
}

// ---------------------------------------------------------------------------
// Phase 0: W [D,D] f32 -> whi/wlo bf16 (row-major, same indexing)
// ---------------------------------------------------------------------------
__global__ void convert_w_kernel(const float* __restrict__ W,
                                 unsigned short* __restrict__ whi,
                                 unsigned short* __restrict__ wlo) {
  const int i = blockIdx.x * blockDim.x + threadIdx.x;  // one per 8 elements
  const float4* p = (const float4*)W;
  float4 a = p[2 * i], b = p[2 * i + 1];
  uint4 hi, lo;
  split8(a, b, hi, lo);
  ((uint4*)whi)[i] = hi;
  ((uint4*)wlo)[i] = lo;
}

// ---------------------------------------------------------------------------
// Phase 1 GEMM: C[m,n] = sum_k X[m,k] * W[n,k] + bias[n]
// Block: 256 threads = 8 waves. Tile: BM=128 (16 rows/wave), BN=64, BK=32.
// B-operand (split bf16) staged in LDS, double buffered, async when possible.
// ---------------------------------------------------------------------------
__launch_bounds__(256, 2)
__global__ void gemm_wx_kernel(const float* __restrict__ X,
                               const unsigned short* __restrict__ whi,
                               const unsigned short* __restrict__ wlo,
                               const float* __restrict__ bias,
                               float* __restrict__ Wx,
                               int N, int K) {
  constexpr int BN = 64, BK = 32;
  constexpr int RS = 5;  // uint4 per LDS row: 80B stride (32 bf16 + 16B pad)
  __shared__ uint4 sBh[2][BN * RS];
  __shared__ uint4 sBl[2][BN * RS];

  const int tid  = threadIdx.x;
  const int wave = tid >> 5;
  const int lane = tid & 31;
  const int l16  = lane & 15;
  const int lhi  = lane >> 4;          // k-half selector
  const int m0   = blockIdx.y * 128 + wave * 16;
  const int n0   = blockIdx.x * BN;

  // cooperative B-tile loader: 4 threads per row, 8 bf16 (16B) each
  const int brow = tid >> 2;           // 0..63
  const int bkc  = (tid & 3) * 8;      // 0,8,16,24
  const size_t wbase = (size_t)(n0 + brow) * K + bkc;
  const int bslot = brow * RS + (bkc >> 3);

  // accumulators, pre-seeded with bias (C/D layout: lane%16 = column)
  v8f acc[4];
#pragma unroll
  for (int nt = 0; nt < 4; ++nt) {
    float bv = bias[n0 + nt * 16 + l16];
#pragma unroll
    for (int r = 0; r < 8; ++r) acc[nt][r] = bv;
  }

  const int KS = K / BK;
  const float* xrow = X + (size_t)(m0 + l16) * K + lhi * 8;

#if HAVE_ASYNC_LDS
  // prime buffer 0 via async DMA (no VGPR round-trip)
  __builtin_amdgcn_global_load_async_to_lds_b128(
      ASYNC_GSRC(whi + wbase), ASYNC_LDST(&sBh[0][bslot]), 0, 0);
  __builtin_amdgcn_global_load_async_to_lds_b128(
      ASYNC_GSRC(wlo + wbase), ASYNC_LDST(&sBl[0][bslot]), 0, 0);
  wait_async0();
  __syncthreads();
#else
  sBh[0][bslot] = *(const uint4*)(whi + wbase);
  sBl[0][bslot] = *(const uint4*)(wlo + wbase);
  __syncthreads();
#endif

  for (int ks = 0; ks < KS; ++ks) {
    const int cur = ks & 1, nxt = cur ^ 1;

#if HAVE_ASYNC_LDS
    if (ks + 1 < KS) {  // kick next tile's DMA before computing on current
      const size_t w = wbase + (size_t)(ks + 1) * BK;
      __builtin_amdgcn_global_load_async_to_lds_b128(
          ASYNC_GSRC(whi + w), ASYNC_LDST(&sBh[nxt][bslot]), 0, 0);
      __builtin_amdgcn_global_load_async_to_lds_b128(
          ASYNC_GSRC(wlo + w), ASYNC_LDST(&sBl[nxt][bslot]), 0, 0);
    }
#else
    uint4 nh, nl;
    if (ks + 1 < KS) {
      const size_t w = wbase + (size_t)(ks + 1) * BK;
      nh = *(const uint4*)(whi + w);
      nl = *(const uint4*)(wlo + w);
    }
#endif

    // A fragment: lane holds rows m0+l16, k-chunks lhi*8 and 16+lhi*8
    const float* xp = xrow + (size_t)ks * BK;
    float4 a0 = *(const float4*)(xp);
    float4 a1 = *(const float4*)(xp + 4);
    float4 a2 = *(const float4*)(xp + 16);
    float4 a3 = *(const float4*)(xp + 20);
    union { v16bf v; uint4 q[2]; } Ah, Al;
    split8(a0, a1, Ah.q[0], Al.q[0]);
    split8(a2, a3, Ah.q[1], Al.q[1]);

#pragma unroll
    for (int nt = 0; nt < 4; ++nt) {
      const int ridx = (nt * 16 + l16) * RS + lhi;
      union { v16bf v; uint4 q[2]; } Bh, Bl;
      Bh.q[0] = sBh[cur][ridx]; Bh.q[1] = sBh[cur][ridx + 2];
      Bl.q[0] = sBl[cur][ridx]; Bl.q[1] = sBl[cur][ridx + 2];
      // split-bf16 fp32 emulation: hi*hi + hi*lo + lo*hi
      acc[nt] = __builtin_amdgcn_wmma_f32_16x16x32_bf16(
          false, Ah.v, false, Bh.v, (short)0, acc[nt], false, false);
      acc[nt] = __builtin_amdgcn_wmma_f32_16x16x32_bf16(
          false, Ah.v, false, Bl.v, (short)0, acc[nt], false, false);
      acc[nt] = __builtin_amdgcn_wmma_f32_16x16x32_bf16(
          false, Al.v, false, Bh.v, (short)0, acc[nt], false, false);
    }

#if HAVE_ASYNC_LDS
    if (ks + 1 < KS) wait_async0();  // own DMAs done; barrier covers the rest
    __syncthreads();
#else
    __syncthreads();
    if (ks + 1 < KS) {
      sBh[nxt][bslot] = nh; sBl[nxt][bslot] = nl;
      __syncthreads();
    }
#endif
  }

  // store: VGPR r -> row m0 + r (+8 for lanes 16..31), col = n0 + nt*16 + l16
  const int rbase = m0 + (lhi ? 8 : 0);
#pragma unroll
  for (int nt = 0; nt < 4; ++nt) {
#pragma unroll
    for (int r = 0; r < 8; ++r) {
      Wx[(size_t)(rbase + r) * N + (n0 + nt * 16 + l16)] = acc[nt][r];
    }
  }
}

// ---------------------------------------------------------------------------
// Phase 2 Scan: one wave32 per batch row; lane keeps 32 h-values in registers.
// Wx for step t lives at hseq[(t+1)*B*D ...] and is overwritten by h_{t+1}.
// ---------------------------------------------------------------------------
__launch_bounds__(32)
__global__ void scan_kernel(const float* __restrict__ h0,
                            const float* __restrict__ log_alpha,
                            float* __restrict__ out,
                            float* __restrict__ hseq,
                            int T, int B, int D) {
  const int b    = blockIdx.x;
  const int lane = threadIdx.x;          // 0..31
  const float alpha = expf(log_alpha[0]);
  const float invD  = 1.0f / (float)D;
  const size_t rowoff = (size_t)b * D;
  const size_t step   = (size_t)B * D;
  const int base = lane * 4;             // chunk j at d = j*128 + lane*4

  float4 h[8];
#pragma unroll
  for (int j = 0; j < 8; ++j)
    h[j] = *(const float4*)(h0 + rowoff + j * 128 + base);
#pragma unroll
  for (int j = 0; j < 8; ++j)            // h[0] = h0
    *(float4*)(hseq + rowoff + j * 128 + base) = h[j];

  // register double-buffer for Wx
  float4 wx[2][8];
#pragma unroll
  for (int j = 0; j < 8; ++j)
    wx[0][j] = *(const float4*)(hseq + step + rowoff + j * 128 + base);

  for (int t = 0; t < T; ++t) {
    const int cur = t & 1, nxt = cur ^ 1;
    if (t + 1 < T) {                     // prefetch next Wx (independent of h)
      const float* p = hseq + (size_t)(t + 2) * step + rowoff;
#pragma unroll
      for (int j = 0; j < 8; ++j)
        wx[nxt][j] = *(const float4*)(p + j * 128 + base);
    }
    if (t + 2 < T) {                     // distance-2 line prefetch into caches
      const float* p2 = hseq + (size_t)(t + 3) * step + rowoff;
#pragma unroll
      for (int j = 0; j < 8; ++j)
        __builtin_prefetch((const void*)(p2 + j * 128 + base), 0, 0);
    }

    float4 hr[8];
    float ss = 0.0f;
#pragma unroll
    for (int j = 0; j < 8; ++j) {
      hr[j].x = fmaf(alpha, wx[cur][j].x, h[j].x);
      hr[j].y = fmaf(alpha, wx[cur][j].y, h[j].y);
      hr[j].z = fmaf(alpha, wx[cur][j].z, h[j].z);
      hr[j].w = fmaf(alpha, wx[cur][j].w, h[j].w);
      ss = fmaf(hr[j].x, hr[j].x, ss);
      ss = fmaf(hr[j].y, hr[j].y, ss);
      ss = fmaf(hr[j].z, hr[j].z, ss);
      ss = fmaf(hr[j].w, hr[j].w, ss);
    }
    // wave32 reduction, no LDS / no barrier
#pragma unroll
    for (int off = 16; off > 0; off >>= 1) ss += __shfl_xor(ss, off, 32);

    const float mean = fmaf(ss, invD, 1e-6f);
    float r = rsqrtf(mean);
    r = r * (1.5f - 0.5f * mean * r * r);  // one Newton step

    float* hp = hseq + (size_t)(t + 1) * step + rowoff;  // overwrites Wx[t]
    float* op = out + (size_t)t * step + rowoff;
#pragma unroll
    for (int j = 0; j < 8; ++j) {
      float4 hn, o;
      hn.x = hr[j].x * r; hn.y = hr[j].y * r;
      hn.z = hr[j].z * r; hn.w = hr[j].w * r;
      o.x = hn.x * hn.x / (1.0f + __expf(-hn.x));
      o.y = hn.y * hn.y / (1.0f + __expf(-hn.y));
      o.z = hn.z * hn.z / (1.0f + __expf(-hn.z));
      o.w = hn.w * hn.w / (1.0f + __expf(-hn.w));
      *(float4*)(hp + j * 128 + base) = hn;
      *(float4*)(op + j * 128 + base) = o;
      h[j] = hn;
    }
  }
}

// ---------------------------------------------------------------------------
extern "C" void kernel_launch(void* const* d_in, const int* in_sizes, int n_in,
                              void* d_out, int out_size, void* d_ws, size_t ws_size,
                              hipStream_t stream) {
  const float* x         = (const float*)d_in[0];  // [T,B,D]
  const float* h0        = (const float*)d_in[1];  // [B,D]
  const float* W         = (const float*)d_in[2];  // [D,D]
  const float* bias      = (const float*)d_in[3];  // [D]
  const float* log_alpha = (const float*)d_in[4];  // scalar

  const int D = in_sizes[3];
  const int B = in_sizes[1] / D;
  const int T = in_sizes[0] / (B * D);
  const int M = T * B;                    // 32768

  float* out  = (float*)d_out;                    // output [T,B,D]
  float* hseq = out + (size_t)T * B * D;          // h [T+1,B,D]
  float* Wx   = hseq + (size_t)B * D;             // Wx staged in h[1..T] slots

  // Split-bf16 W stash lives in the out[] region (overwritten later by scan,
  // which only runs after the GEMM is complete) -> no workspace dependence.
  unsigned short* whi = (unsigned short*)out;     // D*D bf16 (2MB)
  unsigned short* wlo = whi + (size_t)D * D;      // D*D bf16 (2MB)

  // Phase 0: W f32 -> bf16 hi/lo
  convert_w_kernel<<<(D * D / 8 + 255) / 256, 256, 0, stream>>>(W, whi, wlo);

  // Phase 1: Wx = X @ W^T + b  (split-bf16 WMMA; row m = t*B+b -> h[t+1] slot)
  dim3 ggrid(D / 64, M / 128);
  gemm_wx_kernel<<<ggrid, 256, 0, stream>>>(x, whi, wlo, bias, Wx, D, D);

  // Phase 2: sequential scan, one wave per batch row (Wx hot in 192MB L2)
  scan_kernel<<<B, 32, 0, stream>>>(h0, log_alpha, out, hseq, T, B, D);
}
